// Query_Gen_transformer_8005819039904
// MI455X (gfx1250) — matrix operand order repaired and checked by reading, not compiled, not run on hardware
//
#include <hip/hip_runtime.h>
#include <hip/hip_bf16.h>

typedef __attribute__((ext_vector_type(16))) _Float16 v16h;
typedef __attribute__((ext_vector_type(8)))  float    v8f;

#define TOK   65536
#define NSEQ  4096
#define BATCH 16
#define DIMM  256
#define HIDD  512
#define NHEAD 8
#define HD    32

__device__ __forceinline__ v8f vzero8() {
  v8f z = {0.f, 0.f, 0.f, 0.f, 0.f, 0.f, 0.f, 0.f};
  return z;
}

__device__ __forceinline__ v8f wmma16(v16h a, v16h b, v8f c) {
  // emits v_wmma_f32_16x16x32_f16 (D = A*B + C, f32 accum)
  return __builtin_amdgcn_wmma_f32_16x16x32_f16(false, a, false, b, (short)0, c,
                                                false, false);
}

// A-fragment (16x32 f16, MxK) from row-major tile. base already includes k-step
// offset. lane L -> row M=L&15; halves 0..15 at contiguous K=kb+base..+15,
// base = (L>=16)?16:0.
__device__ __forceinline__ v16h load_a(const _Float16* base, int ld) {
  int lane = threadIdx.x & 31;
  const _Float16* p0 = base + (lane & 15) * ld + ((lane >> 4) << 4);
  union { v16h v; _Float16 h[16]; } u;
#pragma unroll
  for (int i = 0; i < 16; ++i) u.h[i] = p0[i];
  return u.v;
}

// B-fragment (32x16 f16, KxN) from an N-major (N x K) tile: lane L -> column
// N=L&15, contiguous 16 K-values starting at ((L>=16)?16:0). base includes kb.
__device__ __forceinline__ v16h load_b(const _Float16* base, int ld) {
  int lane = threadIdx.x & 31;
  const _Float16* p = base + (lane & 15) * ld + ((lane >> 4) << 4);
  union { v16h v; _Float16 h[16]; } u;
#pragma unroll
  for (int i = 0; i < 16; ++i) u.h[i] = p[i];
  return u.v;
}

// ---------- prep: W (K x N, f32) -> WT (N x K, f16) ----------
__global__ void k_convT(const float* __restrict__ W, _Float16* __restrict__ WT,
                        int K, int N) {
  int idx = blockIdx.x * blockDim.x + threadIdx.x;
  if (idx >= K * N) return;
  int n = idx / K, k = idx - n * K;
  WT[idx] = (_Float16)W[k * N + n];
}

// ---------- q = (LN(query_param) @ wq + bq) * (1/sqrt(32)), f16 ----------
__global__ void k_qprep(const float* __restrict__ qp, const float* __restrict__ g,
                        const float* __restrict__ bb, const float* __restrict__ wq,
                        const float* __restrict__ bq, _Float16* __restrict__ qf) {
  __shared__ float row[DIMM], nrm[DIMM], red[DIMM], stat[2];
  int r = blockIdx.x, t = threadIdx.x;
  row[t] = qp[r * DIMM + t];
  __syncthreads();
  red[t] = row[t];
  __syncthreads();
  for (int s = 128; s > 0; s >>= 1) {
    if (t < s) red[t] += red[t + s];
    __syncthreads();
  }
  if (t == 0) stat[0] = red[0] / (float)DIMM;
  __syncthreads();
  float mu = stat[0];
  float d = row[t] - mu;
  red[t] = d * d;
  __syncthreads();
  for (int s = 128; s > 0; s >>= 1) {
    if (t < s) red[t] += red[t + s];
    __syncthreads();
  }
  if (t == 0) stat[1] = rsqrtf(red[0] / (float)DIMM + 1e-5f);
  __syncthreads();
  nrm[t] = (row[t] - mu) * stat[1] * g[t] + bb[t];
  __syncthreads();
  float s = 0.f;
  for (int c = 0; c < DIMM; ++c) s += nrm[c] * wq[c * DIMM + t];
  qf[r * DIMM + t] = (_Float16)((s + bq[t]) * 0.17677669529663687f);
}

// ---------- MLP layer1: h1 = leaky(x*w1 + b1), f16 [TOK,128] ----------
__global__ void k_mlp1(const float* __restrict__ in, int s,
                       const float* __restrict__ w1, const float* __restrict__ b1,
                       _Float16* __restrict__ H1) {
  int idx = blockIdx.x * 256 + threadIdx.x;
  int tok = idx >> 7, j = idx & 127;
  float x = in[tok * 2 + s];
  float h = x * w1[j] + b1[j];
  H1[idx] = (_Float16)(h > 0.f ? h : 0.01f * h);
}

// ---------- MLP layer2: H2 = leaky(H1 @ w2 + b2), WMMA ----------
__global__ void k_mlp2(const _Float16* __restrict__ H1,
                       const _Float16* __restrict__ w2T,
                       const float* __restrict__ b2, _Float16* __restrict__ H2) {
  __shared__ _Float16 At[16 * 128];
  int tok0 = blockIdx.x * 16, tid = threadIdx.x;
  {
    const uint32_t* src = (const uint32_t*)(H1 + (size_t)tok0 * 128);
    uint32_t* dst = (uint32_t*)At;
    for (int i = tid; i < 1024; i += 256) dst[i] = src[i];
  }
  __syncthreads();
  int wave = tid >> 5, lane = tid & 31;
  int Mofs = (lane >> 4) << 3, Nc = lane & 15;
  v8f acc[4] = {vzero8(), vzero8(), vzero8(), vzero8()};
  for (int ks = 0; ks < 4; ++ks) {
    int kb = ks * 32;
    v16h a = load_a(At + kb, 128);
#pragma unroll
    for (int t = 0; t < 4; ++t) {
      int n0 = wave * 64 + t * 16;
      acc[t] = wmma16(a, load_b(w2T + n0 * 128 + kb, 128), acc[t]);
    }
  }
#pragma unroll
  for (int t = 0; t < 4; ++t) {
    int n0 = wave * 64 + t * 16;
#pragma unroll
    for (int r = 0; r < 8; ++r) {
      float v = acc[t][r] + b2[n0 + Nc];
      v = v > 0.f ? v : 0.01f * v;
      H2[(size_t)(tok0 + r + Mofs) * 512 + n0 + Nc] = (_Float16)v;
    }
  }
}

// ---------- fused: t = H2@w3+b3 ; LN(t) ; out = LN @ wp + bp (f16) ----------
__global__ void k_mlp3(const _Float16* __restrict__ H2,
                       const _Float16* __restrict__ w3T,
                       const float* __restrict__ b3, const float* __restrict__ g,
                       const float* __restrict__ be,
                       const _Float16* __restrict__ wpT,
                       const float* __restrict__ bp, _Float16* __restrict__ outp) {
  __shared__ _Float16 A2[16 * 512];
  __shared__ float T[16 * 512];
  __shared__ float redS[16][16], redQ[16][16];
  __shared__ float mu[16], rs[16];
  int tok0 = blockIdx.x * 16, tid = threadIdx.x;
  {
    const uint32_t* src = (const uint32_t*)(H2 + (size_t)tok0 * 512);
    uint32_t* dst = (uint32_t*)A2;
    for (int i = tid; i < 4096; i += 256) dst[i] = src[i];
  }
  __syncthreads();
  int wave = tid >> 5, lane = tid & 31;
  int Mofs = (lane >> 4) << 3, Nc = lane & 15;
  {
    v8f acc[4] = {vzero8(), vzero8(), vzero8(), vzero8()};
    for (int ks = 0; ks < 16; ++ks) {
      int kb = ks * 32;
      v16h a = load_a(A2 + kb, 512);
#pragma unroll
      for (int t = 0; t < 4; ++t) {
        int n0 = wave * 64 + t * 16;
        acc[t] = wmma16(a, load_b(w3T + n0 * 512 + kb, 512), acc[t]);
      }
    }
#pragma unroll
    for (int t = 0; t < 4; ++t) {
      int n0 = wave * 64 + t * 16;
#pragma unroll
      for (int r = 0; r < 8; ++r)
        T[(r + Mofs) * 512 + n0 + Nc] = acc[t][r] + b3[n0 + Nc];
    }
  }
  __syncthreads();
  // LayerNorm over 512, 16 threads per row
  int rr = tid >> 4, c0 = tid & 15;
  float s = 0.f, s2 = 0.f;
  for (int j = c0; j < 512; j += 16) { float v = T[rr * 512 + j]; s += v; s2 += v * v; }
  redS[rr][c0] = s; redQ[rr][c0] = s2;
  __syncthreads();
  if (c0 == 0) {
    float ts = 0.f, tq = 0.f;
    for (int i = 0; i < 16; ++i) { ts += redS[rr][i]; tq += redQ[rr][i]; }
    float m = ts / 512.f;
    mu[rr] = m;
    rs[rr] = rsqrtf(tq / 512.f - m * m + 1e-5f);
  }
  __syncthreads();
  for (int j = c0; j < 512; j += 16) {
    float v = (T[rr * 512 + j] - mu[rr]) * rs[rr] * g[j] + be[j];
    A2[rr * 512 + j] = (_Float16)v;  // reuse A2 as LN output
  }
  __syncthreads();
  // projection 512 -> 256
  v8f acc2[2] = {vzero8(), vzero8()};
  for (int ks = 0; ks < 16; ++ks) {
    int kb = ks * 32;
    v16h a = load_a(A2 + kb, 512);
#pragma unroll
    for (int t = 0; t < 2; ++t) {
      int n0 = wave * 32 + t * 16;
      acc2[t] = wmma16(a, load_b(wpT + n0 * 512 + kb, 512), acc2[t]);
    }
  }
#pragma unroll
  for (int t = 0; t < 2; ++t) {
    int n0 = wave * 32 + t * 16;
#pragma unroll
    for (int r = 0; r < 8; ++r)
      outp[(size_t)(tok0 + r + Mofs) * 256 + n0 + Nc] =
          (_Float16)(acc2[t][r] + bp[n0 + Nc]);
  }
}

// ---------- flash attention, one block per (batch, head) ----------
__global__ void k_attn(const _Float16* __restrict__ qf,
                       const _Float16* __restrict__ kf,
                       const _Float16* __restrict__ vf, float* __restrict__ ao) {
  __shared__ _Float16 Kt[32 * 32];       // [key][d]
  __shared__ _Float16 Vt[32 * 32];       // [d][key]
  __shared__ _Float16 Pb[8][16 * 32];    // per-wave P staging
  int b = blockIdx.x >> 3, h = blockIdx.x & 7;
  int tid = threadIdx.x, wave = tid >> 5, lane = tid & 31;
  int Mofs = (lane >> 4) << 3, Nc = lane & 15;
  v16h aq[2];
#pragma unroll
  for (int t = 0; t < 2; ++t)
    aq[t] = load_a(qf + (size_t)(wave * 32 + t * 16) * 256 + h * 32, 256);
  float m[2][8], l[2][8];
  v8f o[2][2] = {{vzero8(), vzero8()}, {vzero8(), vzero8()}};
#pragma unroll
  for (int t = 0; t < 2; ++t)
#pragma unroll
    for (int r = 0; r < 8; ++r) { m[t][r] = -1e30f; l[t][r] = 0.f; }
  const _Float16* kbp = kf + (size_t)b * NSEQ * 256 + h * 32;
  const _Float16* vbp = vf + (size_t)b * NSEQ * 256 + h * 32;

  for (int kc = 0; kc < NSEQ / 32; ++kc) {
#pragma unroll
    for (int j = 0; j < 4; ++j) {
      int e = tid * 4 + j;
      int key = e >> 5, d = e & 31;
      _Float16 kvk = kbp[(size_t)(kc * 32 + key) * 256 + d];
      _Float16 kvv = vbp[(size_t)(kc * 32 + key) * 256 + d];
      Kt[e] = kvk;
      Vt[d * 32 + key] = kvv;
    }
    __syncthreads();
#pragma unroll
    for (int t = 0; t < 2; ++t) {
      v8f s0 = wmma16(aq[t], load_b(Kt, 32), vzero8());
      v8f s1 = wmma16(aq[t], load_b(Kt + 16 * 32, 32), vzero8());
#pragma unroll
      for (int r = 0; r < 8; ++r) {
        float mx = fmaxf(s0[r], s1[r]);
#pragma unroll
        for (int sft = 1; sft < 16; sft <<= 1) mx = fmaxf(mx, __shfl_xor(mx, sft, 32));
        float mnew = fmaxf(m[t][r], mx);
        float alpha = __expf(m[t][r] - mnew);
        float p0 = __expf(s0[r] - mnew);
        float p1 = __expf(s1[r] - mnew);
        float rsum = p0 + p1;
#pragma unroll
        for (int sft = 1; sft < 16; sft <<= 1) rsum += __shfl_xor(rsum, sft, 32);
        l[t][r] = l[t][r] * alpha + rsum;
        m[t][r] = mnew;
        o[t][0][r] *= alpha;
        o[t][1][r] *= alpha;
        int M = r + Mofs;
        Pb[wave][M * 32 + Nc] = (_Float16)p0;
        Pb[wave][M * 32 + 16 + Nc] = (_Float16)p1;
      }
      __builtin_amdgcn_wave_barrier();  // order per-wave LDS store->load
      v16h ap = load_a(Pb[wave], 32);
      o[t][0] = wmma16(ap, load_b(Vt, 32), o[t][0]);
      o[t][1] = wmma16(ap, load_b(Vt + 16 * 32, 32), o[t][1]);
    }
    __syncthreads();
  }
#pragma unroll
  for (int t = 0; t < 2; ++t)
#pragma unroll
    for (int dt = 0; dt < 2; ++dt)
#pragma unroll
      for (int r = 0; r < 8; ++r) {
        int row = wave * 32 + t * 16 + r + Mofs;
        ao[((size_t)b * 256 + row) * 256 + h * 32 + dt * 16 + Nc] =
            o[t][dt][r] / l[t][r];
      }
}

// ---------- final: out = attn @ wo + bo (f32 out) ----------
__global__ void k_oproj(const float* __restrict__ ao,
                        const _Float16* __restrict__ woT,
                        const float* __restrict__ bo, float* __restrict__ out) {
  __shared__ _Float16 Af[16 * 256];
  int row0 = blockIdx.x * 16, tid = threadIdx.x;
  for (int i = tid; i < 4096; i += 256) Af[i] = (_Float16)ao[(size_t)row0 * 256 + i];
  __syncthreads();
  int wave = tid >> 5, lane = tid & 31;
  int Mofs = (lane >> 4) << 3, Nc = lane & 15;
  v8f acc[2] = {vzero8(), vzero8()};
  for (int ks = 0; ks < 8; ++ks) {
    int kb = ks * 32;
    v16h a = load_a(Af + kb, 256);
#pragma unroll
    for (int t = 0; t < 2; ++t) {
      int n0 = wave * 32 + t * 16;
      acc[t] = wmma16(a, load_b(woT + n0 * 256 + kb, 256), acc[t]);
    }
  }
#pragma unroll
  for (int t = 0; t < 2; ++t) {
    int n0 = wave * 32 + t * 16;
#pragma unroll
    for (int r = 0; r < 8; ++r)
      out[(size_t)(row0 + r + Mofs) * 256 + n0 + Nc] = acc[t][r] + bo[n0 + Nc];
  }
}

extern "C" void kernel_launch(void* const* d_in, const int* in_sizes, int n_in,
                              void* d_out, int out_size, void* d_ws, size_t ws_size,
                              hipStream_t stream) {
  const float* input = (const float*)d_in[0];
  const float* qp    = (const float*)d_in[1];
  const float* w1    = (const float*)d_in[2];
  const float* b1    = (const float*)d_in[3];
  const float* w2    = (const float*)d_in[4];
  const float* b2    = (const float*)d_in[5];
  const float* w3    = (const float*)d_in[6];
  const float* b3    = (const float*)d_in[7];
  const float* lnqg  = (const float*)d_in[8];
  const float* lnqb  = (const float*)d_in[9];
  const float* lnkg  = (const float*)d_in[10];
  const float* lnkb  = (const float*)d_in[11];
  const float* lnvg  = (const float*)d_in[12];
  const float* lnvb  = (const float*)d_in[13];
  const float* wq    = (const float*)d_in[14];
  const float* bq    = (const float*)d_in[15];
  const float* wk    = (const float*)d_in[16];
  const float* bk    = (const float*)d_in[17];
  const float* wv    = (const float*)d_in[18];
  const float* bv    = (const float*)d_in[19];
  const float* wo    = (const float*)d_in[20];
  const float* bo    = (const float*)d_in[21];

  char* ws = (char*)d_ws;
  size_t off = 0;
  auto take = [&](size_t bytes) -> char* {
    char* p = ws + off;
    off = (off + bytes + 255) & ~(size_t)255;
    return p;
  };
  _Float16* w2T = (_Float16*)take((size_t)512 * 128 * 2);
  _Float16* w3T = (_Float16*)take((size_t)512 * 512 * 2);
  _Float16* wkT = (_Float16*)take((size_t)256 * 512 * 2);
  _Float16* wvT = (_Float16*)take((size_t)256 * 512 * 2);
  _Float16* woT = (_Float16*)take((size_t)256 * 256 * 2);
  _Float16* qfh = (_Float16*)take((size_t)256 * 256 * 2);
  _Float16* H1  = (_Float16*)take((size_t)TOK * 128 * 2);
  _Float16* H2  = (_Float16*)take((size_t)TOK * 512 * 2);
  _Float16* kfh = (_Float16*)take((size_t)TOK * 256 * 2);
  _Float16* vfh = (_Float16*)take((size_t)TOK * 256 * 2);
  float*    ao  = (float*)take((size_t)BATCH * 256 * 256 * 4);

  k_convT<<<(512 * 128) / 256, 256, 0, stream>>>(w2, w2T, 128, 512);
  k_convT<<<(512 * 512) / 256, 256, 0, stream>>>(w3, w3T, 512, 512);
  k_convT<<<(512 * 256) / 256, 256, 0, stream>>>(wk, wkT, 512, 256);
  k_convT<<<(512 * 256) / 256, 256, 0, stream>>>(wv, wvT, 512, 256);
  k_convT<<<(256 * 256) / 256, 256, 0, stream>>>(wo, woT, 256, 256);
  k_qprep<<<256, 256, 0, stream>>>(qp, lnqg, lnqb, wq, bq, qfh);

  // K path (stream 0 = X)
  k_mlp1<<<(TOK * 128) / 256, 256, 0, stream>>>(input, 0, w1, b1, H1);
  k_mlp2<<<TOK / 16, 256, 0, stream>>>(H1, w2T, b2, H2);
  k_mlp3<<<TOK / 16, 256, 0, stream>>>(H2, w3T, b3, lnkg, lnkb, wkT, bk, kfh);
  // V path (stream 1 = Y)
  k_mlp1<<<(TOK * 128) / 256, 256, 0, stream>>>(input, 1, w1, b1, H1);
  k_mlp2<<<TOK / 16, 256, 0, stream>>>(H1, w2T, b2, H2);
  k_mlp3<<<TOK / 16, 256, 0, stream>>>(H2, w3T, b3, lnvg, lnvb, wvT, bv, vfh);

  k_attn<<<BATCH * NHEAD, 256, 0, stream>>>(qfh, kfh, vfh, ao);
  k_oproj<<<(BATCH * 256) / 16, 256, 0, stream>>>(ao, woT, bo, (float*)d_out);
}
